// RoPEAttention_78245714199297
// MI455X (gfx1250) — matrix operand order repaired and checked by reading, not compiled, not run on hardware
//
#include <hip/hip_runtime.h>
#include <hip/hip_bf16.h>
#include <stdint.h>

// ---------------------------------------------------------------------------
// RoPE attention for MI455X (gfx1250).  All matmuls (QKV proj, Q*K^T, P*V,
// out proj) run on v_wmma_f32_16x16x32_f16 (f16 in, f32 accumulate).
// Shapes: B=4, S=1024, D=1024, H=16, dk=64.
// ---------------------------------------------------------------------------

typedef __attribute__((ext_vector_type(16))) _Float16 v16h;
typedef __attribute__((ext_vector_type(2)))  __fp16   v2fp;
typedef __attribute__((ext_vector_type(8)))  float    v8f;

union FragA { v16h h; uint32_t u[8]; };
union H2U   { v2fp h; uint32_t u; };

__device__ __forceinline__ v8f wmma16(v16h a, v16h b, v8f c) {
  // 8-arg form: (neg_a, A, neg_b, B, c_mod, C, reuse_a, reuse_b)
  return __builtin_amdgcn_wmma_f32_16x16x32_f16(false, a, false, b,
                                                (short)0, c, false, false);
}

// packed f32x2 -> f16x2 (v_cvt_pk_rtz_f16_f32), one VALU op per 2 elements
__device__ __forceinline__ uint32_t pk_f16(float x, float y) {
  H2U r; r.h = __builtin_amdgcn_cvt_pkrtz(x, y); return r.u;
}

#define NB   4
#define NS   1024
#define ND   1024
#define NH   16
#define NDK  64
#define LDA  40   // LDS row stride in halves (80B: 16B-aligned rows)

// ---------------------------------------------------------------------------
// Generic GEMM + bias:  C[4096x1024] = A[4096x1024] * W[1024x1024] + bias
// Block = 128 threads (4 waves), block tile 64x64, wave tile 32x32 (2x2 WMMA).
// A loaded as float4 (b128) + packed-cvt + b64 LDS stores; W loaded as float4
// and scattered k-transposed into LDS so both fragment types are contiguous
// dword gathers in the same k order (k = 16*half + s) for A and B.
// ---------------------------------------------------------------------------
template<bool AHALF>
__global__ __launch_bounds__(128)
void gemm_bias_kernel(const void* __restrict__ Ain,
                      const float* __restrict__ W,
                      const float* __restrict__ bias,
                      float* __restrict__ C)
{
  const int N = ND, K = ND;
  __shared__ __align__(16) _Float16 As[64][LDA];   // [m][k]
  __shared__ __align__(16) _Float16 Bs[64][LDA];   // [n][k]  (k-transposed W)

  const int tid  = threadIdx.x;
  const int wave = tid >> 5, lane = tid & 31;
  const int lr   = lane & 15, hi = lane >> 4;     // 16-lane half of the wave
  const int wr   = wave >> 1, wc = wave & 1;      // 2x2 wave grid in block
  const int m0   = blockIdx.y * 64, n0 = blockIdx.x * 64;

  v8f zero = {};
  v8f acc[2][2] = {{zero, zero}, {zero, zero}};

  for (int kt = 0; kt < K / 32; ++kt) {
    const int k0 = kt * 32;
    __syncthreads();
    // cooperative tile load: 512 float4 (or uint2) each of A and W tiles,
    // 4 vector loads per thread per tile.
    #pragma unroll
    for (int i = 0; i < 4; ++i) {
      int idx = tid + i * 128;            // 0..511
      int ar = idx >> 3, ac = (idx & 7) * 4;       // A: row-major, 4 elems
      if (AHALF) {
        const uint2 av = *(const uint2*)((const _Float16*)Ain
                             + (size_t)(m0 + ar) * K + k0 + ac);
        *(uint2*)&As[ar][ac] = av;
      } else {
        const float4 av = *(const float4*)((const float*)Ain
                             + (size_t)(m0 + ar) * K + k0 + ac);
        uint2 p; p.x = pk_f16(av.x, av.y); p.y = pk_f16(av.z, av.w);
        *(uint2*)&As[ar][ac] = p;
      }
      int bk = idx >> 4, bn = (idx & 15) * 4;      // W: coalesced float4 on n
      const float4 wv = *(const float4*)(W + (size_t)(k0 + bk) * N + n0 + bn);
      Bs[bn + 0][bk] = (_Float16)wv.x;             // k-transposed scatter
      Bs[bn + 1][bk] = (_Float16)wv.y;
      Bs[bn + 2][bk] = (_Float16)wv.z;
      Bs[bn + 3][bk] = (_Float16)wv.w;
    }
    if (kt + 1 < K / 32)                  // speculative prefetch of next W tile
      __builtin_prefetch((const char*)(W + (size_t)(k0 + 32) * N + n0), 0, 0);
    __syncthreads();

    // A fragment: contiguous k order shared with B (lanes<16: K 0-15, else 16-31)
    FragA a[2], b[2];
    #pragma unroll
    for (int mr = 0; mr < 2; ++mr) {
      const _Float16* p = &As[wr * 32 + mr * 16 + lr][0];
      const int kb = hi * 16;
      #pragma unroll
      for (int j = 0; j < 8; ++j)
        a[mr].u[j] = *(const uint32_t*)(p + kb + 2 * j);
    }
    // B fragment: contiguous k order matching A (lane = column, lanes<16: K 0-15, else 16-31)
    #pragma unroll
    for (int nc = 0; nc < 2; ++nc) {
      const _Float16* p = &Bs[wc * 32 + nc * 16 + lr][0];
      const int kb = hi * 16;
      #pragma unroll
      for (int j = 0; j < 8; ++j)
        b[nc].u[j] = *(const uint32_t*)(p + kb + 2 * j);
    }
    acc[0][0] = wmma16(a[0].h, b[0].h, acc[0][0]);
    acc[0][1] = wmma16(a[0].h, b[1].h, acc[0][1]);
    acc[1][0] = wmma16(a[1].h, b[0].h, acc[1][0]);
    acc[1][1] = wmma16(a[1].h, b[1].h, acc[1][1]);
  }

  // C/D layout: VGPR r -> row r (lanes 0-15) / row 8+r (lanes 16-31), col = lane%16
  #pragma unroll
  for (int mr = 0; mr < 2; ++mr)
    #pragma unroll
    for (int nc = 0; nc < 2; ++nc)
      #pragma unroll
      for (int r = 0; r < 8; ++r) {
        int row = m0 + wr * 32 + mr * 16 + hi * 8 + r;
        int col = n0 + wc * 32 + nc * 16 + lr;
        C[(size_t)row * N + col] = acc[mr][nc][r] + bias[col];
      }
}

// ---------------------------------------------------------------------------
// RoPE + layout change.  One thread per (b,s,h,pair).
//   qh,kh: f16 [B,H,S,dk]   (A / B^T friendly for Q*K^T)
//   vt   : f16 [B,H,dk,S]   (B friendly for P*V)
// ---------------------------------------------------------------------------
__global__ __launch_bounds__(256)
void rope_transpose_kernel(const float* __restrict__ qp,
                           const float* __restrict__ kp,
                           const float* __restrict__ vp,
                           _Float16* __restrict__ qh,
                           _Float16* __restrict__ kh,
                           _Float16* __restrict__ vt)
{
  int idx = blockIdx.x * blockDim.x + threadIdx.x;   // B*S*H*32 = 2,097,152
  if (idx >= NB * NS * NH * (NDK / 2)) return;
  int i = idx & 31;            // rotary pair index (0..31)
  int h = (idx >> 5) & 15;
  int s = (idx >> 9) & 1023;
  int b = idx >> 19;

  // inv_freq = 10000^(-2i/64) = exp(-(2i/64)*ln(10000))
  float inv_freq = __expf(-(float)(2 * i) * (9.210340372f / 64.0f));
  float ang = (float)s * inv_freq;
  float c, sn;
  __sincosf(ang, &sn, &c);     // sn = sin, c = cos
  size_t src = (((size_t)b * NS + s) * ND) + h * NDK + 2 * i;
  size_t dst = ((((size_t)b * NH + h) * NS) + s) * NDK + 2 * i;

  float q0 = qp[src], q1 = qp[src + 1];
  qh[dst]     = (_Float16)(q0 * c - q1 * sn);
  qh[dst + 1] = (_Float16)(q0 * sn + q1 * c);

  float k0 = kp[src], k1 = kp[src + 1];
  kh[dst]     = (_Float16)(k0 * c - k1 * sn);
  kh[dst + 1] = (_Float16)(k0 * sn + k1 * c);

  size_t vd = ((((size_t)b * NH + h) * NDK) + 2 * i) * NS + s;
  vt[vd]      = (_Float16)vp[src];
  vt[vd + NS] = (_Float16)vp[src + 1];
}

// ---------------------------------------------------------------------------
// Flash-style attention.  grid = (B*H, S/64), block = 128 (4 waves).
// Each wave: 16 queries x dk=64.  Per 32-key block:
//   4 WMMA for scores, shfl-xor online softmax, P transposed C->A layout via
//   per-wave LDS bounce (s_wait_dscnt), 4 WMMA for P*V.
// Writes f16 context in [B,S,D] for the output GEMM.
// ---------------------------------------------------------------------------
__global__ __launch_bounds__(128)
void attention_kernel(const _Float16* __restrict__ qh,
                      const _Float16* __restrict__ kh,
                      const _Float16* __restrict__ vt,
                      _Float16* __restrict__ ctx)
{
  __shared__ _Float16 Ps[4][16][36];   // per-wave P bounce buffer

  const int tid  = threadIdx.x;
  const int wave = tid >> 5, lane = tid & 31;
  const int lr   = lane & 15, hi = lane >> 4;
  const int bh   = blockIdx.x;              // 0..63
  const int b    = bh >> 4, h = bh & 15;
  const int q0r  = blockIdx.y * 64 + wave * 16;

  const _Float16* qbase = qh + ((size_t)bh * NS + q0r) * NDK;
  const _Float16* kbase = kh + (size_t)bh * NS * NDK;
  const _Float16* vbase = vt + (size_t)bh * NDK * NS;

  // Q A-fragments for dk 0..31 and 32..63 (loaded once, straight from global)
  FragA qa[2];
  {
    const _Float16* p = qbase + (size_t)lr * NDK;
    const int kb = hi * 16;
    #pragma unroll
    for (int f = 0; f < 2; ++f)
      #pragma unroll
      for (int j = 0; j < 8; ++j)
        qa[f].u[j] = *(const uint32_t*)(p + f * 32 + kb + 2 * j);
  }

  v8f zero = {};
  v8f o[4] = {zero, zero, zero, zero};      // 16 x 64 f32 context accumulator
  float m[8], l[8];
  #pragma unroll
  for (int r = 0; r < 8; ++r) { m[r] = -1e30f; l[r] = 0.0f; }

  for (int kb32 = 0; kb32 < NS / 32; ++kb32) {
    const int k0 = kb32 * 32;

    // K^T B-fragments: lane = key column; contiguous 32B loads of dk halves
    v16h kf[2][2];
    #pragma unroll
    for (int kt = 0; kt < 2; ++kt)
      #pragma unroll
      for (int kc = 0; kc < 2; ++kc)
        kf[kt][kc] = *(const v16h*)(kbase + (size_t)(k0 + kt * 16 + lr) * NDK
                                          + kc * 32 + hi * 16);

    v8f s0 = zero, s1 = zero;
    s0 = wmma16(qa[0].h, kf[0][0], s0);
    s0 = wmma16(qa[1].h, kf[0][1], s0);
    s1 = wmma16(qa[0].h, kf[1][0], s1);
    s1 = wmma16(qa[1].h, kf[1][1], s1);

    // online softmax (rows live in VGPR index, keys across the 16-lane group)
    float alpha[8];
    #pragma unroll
    for (int r = 0; r < 8; ++r) {
      float sc0 = s0[r] * 0.125f, sc1 = s1[r] * 0.125f;  // 1/sqrt(64)
      float t = fmaxf(sc0, sc1);
      t = fmaxf(t, __shfl_xor(t, 1, 32));
      t = fmaxf(t, __shfl_xor(t, 2, 32));
      t = fmaxf(t, __shfl_xor(t, 4, 32));
      t = fmaxf(t, __shfl_xor(t, 8, 32));
      float nm = fmaxf(m[r], t);
      float p0 = __expf(sc0 - nm);
      float p1 = __expf(sc1 - nm);
      alpha[r] = __expf(m[r] - nm);
      m[r] = nm;
      float rs = p0 + p1;
      rs += __shfl_xor(rs, 1, 32);
      rs += __shfl_xor(rs, 2, 32);
      rs += __shfl_xor(rs, 4, 32);
      rs += __shfl_xor(rs, 8, 32);
      l[r] = l[r] * alpha[r] + rs;
      // spill P (C layout) to LDS so it can be re-read in A layout
      int row = hi * 8 + r;
      Ps[wave][row][lr]      = (_Float16)p0;
      Ps[wave][row][16 + lr] = (_Float16)p1;
    }
    // same-wave LDS RAW: wait on DS counter only (no barrier needed)
    asm volatile("s_wait_dscnt 0" ::: "memory");

    FragA pa;
    {
      const _Float16* p = &Ps[wave][lr][0];
      const int kb = hi * 16;
      #pragma unroll
      for (int j = 0; j < 8; ++j)
        pa.u[j] = *(const uint32_t*)(p + kb + 2 * j);
    }

    // rescale running context, then accumulate P*V (V^T layout -> contiguous B frags)
    #pragma unroll
    for (int nt = 0; nt < 4; ++nt)
      #pragma unroll
      for (int r = 0; r < 8; ++r)
        o[nt][r] *= alpha[r];
    #pragma unroll
    for (int nt = 0; nt < 4; ++nt) {
      v16h vf = *(const v16h*)(vbase + (size_t)(nt * 16 + lr) * NS + k0 + hi * 16);
      o[nt] = wmma16(pa.h, vf, o[nt]);
    }
  }

  // normalize and write f16 context in [B,S,D]
  #pragma unroll
  for (int nt = 0; nt < 4; ++nt)
    #pragma unroll
    for (int r = 0; r < 8; ++r) {
      int row = q0r + hi * 8 + r;
      int col = h * NDK + nt * 16 + lr;
      ctx[((size_t)b * NS + row) * ND + col] = (_Float16)(o[nt][r] / l[r]);
    }
}

// ---------------------------------------------------------------------------
// Host-side launch
// ---------------------------------------------------------------------------
extern "C" void kernel_launch(void* const* d_in, const int* in_sizes, int n_in,
                              void* d_out, int out_size, void* d_ws, size_t ws_size,
                              hipStream_t stream) {
  (void)in_sizes; (void)n_in; (void)out_size; (void)ws_size;
  const float* q  = (const float*)d_in[0];
  const float* k  = (const float*)d_in[1];
  const float* v  = (const float*)d_in[2];
  const float* Wq = (const float*)d_in[3];
  const float* bq = (const float*)d_in[4];
  const float* Wk = (const float*)d_in[5];
  const float* bk = (const float*)d_in[6];
  const float* Wv = (const float*)d_in[7];
  const float* bv = (const float*)d_in[8];
  const float* Wo = (const float*)d_in[9];
  const float* bo = (const float*)d_in[10];
  float* out = (float*)d_out;

  const size_t F32MAT = (size_t)NB * NS * ND * sizeof(float);     // 16.78 MB
  const size_t F16MAT = (size_t)NB * NS * ND * sizeof(uint16_t);  //  8.39 MB
  char* ws = (char*)d_ws;
  float*     qp  = (float*)(ws);
  float*     kp  = (float*)(ws + F32MAT);
  float*     vp  = (float*)(ws + 2 * F32MAT);
  _Float16*  qhp = (_Float16*)(ws + 3 * F32MAT);
  _Float16*  khp = (_Float16*)(ws + 3 * F32MAT + F16MAT);
  _Float16*  vtp = (_Float16*)(ws + 3 * F32MAT + 2 * F16MAT);
  _Float16*  ctx = (_Float16*)(ws + 3 * F32MAT + 3 * F16MAT);

  dim3 gGemm(ND / 64, (NB * NS) / 64);   // (16, 64)
  gemm_bias_kernel<false><<<gGemm, 128, 0, stream>>>(q, Wq, bq, qp);
  gemm_bias_kernel<false><<<gGemm, 128, 0, stream>>>(k, Wk, bk, kp);
  gemm_bias_kernel<false><<<gGemm, 128, 0, stream>>>(v, Wv, bv, vp);

  int nrope = NB * NS * NH * (NDK / 2);
  rope_transpose_kernel<<<(nrope + 255) / 256, 256, 0, stream>>>(qp, kp, vp, qhp, khp, vtp);

  attention_kernel<<<dim3(NB * NH, NS / 64), 128, 0, stream>>>(qhp, khp, vtp, ctx);

  gemm_bias_kernel<true><<<gGemm, 128, 0, stream>>>(ctx, Wo, bo, out);
}
